// H2_88098369176173
// MI455X (gfx1250) — compile-verified, matrix operand order checked by reading, not run on hardware
//
#include <hip/hip_runtime.h>
#include <hip/hip_bf16.h>
#include <math.h>

typedef __attribute__((ext_vector_type(16))) __bf16 v16bf;
typedef __attribute__((ext_vector_type(8)))  float  v8f;

static __device__ __forceinline__ float atomicMaxFloat(float* addr, float val) {
  // ordered-int trick: works when buffer initialized to -inf
  if (val >= 0.f)
    return __int_as_float(atomicMax((int*)addr, __float_as_int(val)));
  else
    return __uint_as_float(atomicMin((unsigned int*)addr, __float_as_uint(val)));
}

// ---------------- generic utility kernels ----------------

__global__ void k_fill(float* __restrict__ p, float v, size_t n) {
  size_t i = (size_t)blockIdx.x * blockDim.x + threadIdx.x;
  if (i < n) p[i] = v;
}

__global__ void k_relu(float* __restrict__ p, size_t n) {
  size_t i = (size_t)blockIdx.x * blockDim.x + threadIdx.x;
  if (i < n) p[i] = fmaxf(p[i], 0.f);
}

__global__ void k_fix_neginf(float* __restrict__ p, size_t n) {
  size_t i = (size_t)blockIdx.x * blockDim.x + threadIdx.x;
  if (i < n) { float v = p[i]; p[i] = (v == -__builtin_inff()) ? 0.f : v; }
}

// dst = (1+eps)*src   (eps is a device scalar)
__global__ void k_scale_copy(float* __restrict__ d, const float* __restrict__ s,
                             const float* __restrict__ eps, size_t n) {
  size_t i = (size_t)blockIdx.x * blockDim.x + threadIdx.x;
  if (i < n) d[i] = (1.f + eps[0]) * s[i];
}

// W [K,N] f32 -> WT [N,K] bf16
__global__ void k_transpose_bf16(const float* __restrict__ W, __bf16* __restrict__ WT,
                                 int K, int N) {
  int i = blockIdx.x * blockDim.x + threadIdx.x;
  if (i >= K * N) return;
  int k = i / N, n = i % N;
  WT[(size_t)n * K + k] = (__bf16)W[i];
}

// ---------------- WMMA GEMM: C[M,128] = A[M,K] @ W[K,128] (+bias)(+relu) ----------------
// WT is W transposed [128 rows, K] in bf16. One wave -> 16 rows x 128 cols (8 WMMA tiles).
__global__ __launch_bounds__(256)
void k_gemm_wmma(const float* __restrict__ A, int lda,
                 const __bf16* __restrict__ WT,
                 const float* __restrict__ bias,
                 float* __restrict__ C, int ldc,
                 int M, int K, int relu) {
  int wave = (int)((blockIdx.x * blockDim.x + threadIdx.x) >> 5);
  int lane = threadIdx.x & 31;
  int row0 = wave * 16;
  if (row0 >= M) return;                 // uniform per wave
  int half = lane >> 4;
  int l16  = lane & 15;
  int arow = row0 + l16;
  if (arow >= M) arow = M - 1;           // clamp: keep EXEC all-ones for WMMA

  v8f acc[8] = {};                       // 8 tiles of 16x16 f32

  for (int k0 = 0; k0 < K; k0 += 32) {
    // A fragment (16x32 bf16): lane holds row M=l16, K = k0+half*8+[0..7] and k0+16+half*8+[0..7]
    const float* ap = A + (size_t)arow * lda + k0 + half * 8;
    v16bf afrag;
#pragma unroll
    for (int i = 0; i < 8; ++i) afrag[i]     = (__bf16)ap[i];
#pragma unroll
    for (int i = 0; i < 8; ++i) afrag[8 + i] = (__bf16)ap[16 + i];

#pragma unroll
    for (int n = 0; n < 8; ++n) {
      // B fragment (32x16 bf16): lane holds column N = n*16+l16, same K pattern -> contiguous in WT
      const __bf16* bp = WT + (size_t)(n * 16 + l16) * K + k0 + half * 8;
      v16bf bfrag;
#pragma unroll
      for (int i = 0; i < 8; ++i) bfrag[i]     = bp[i];
#pragma unroll
      for (int i = 0; i < 8; ++i) bfrag[8 + i] = bp[16 + i];
      acc[n] = __builtin_amdgcn_wmma_f32_16x16x32_bf16(
          false, afrag, false, bfrag, (short)0, acc[n], false, false);
    }
  }

  // D layout: vgpr r, lanes 0-15 -> M=r, lanes 16-31 -> M=8+r; N = l16 within tile
  int crow = row0 + half * 8;
#pragma unroll
  for (int n = 0; n < 8; ++n) {
    int col = n * 16 + l16;
    float b = bias ? bias[col] : 0.f;
#pragma unroll
    for (int r = 0; r < 8; ++r) {
      int rr = crow + r;
      if (rr < M) {
        float v = acc[n][r] + b;
        if (relu) v = fmaxf(v, 0.f);
        C[(size_t)rr * ldc + col] = v;
      }
    }
  }
}

// ---------------- BatchNorm (column stats over M rows, C=128) ----------------

__global__ void k_bn_stats(const float* __restrict__ x, float* __restrict__ sums,
                           float* __restrict__ sumsq, int M) {
  int col = threadIdx.x;                 // blockDim.x == 128
  float s = 0.f, q = 0.f;
  for (int r = blockIdx.x; r < M; r += gridDim.x) {
    float v = x[(size_t)r * 128 + col];
    s += v; q += v * v;
  }
  atomicAdd(&sums[col], s);
  atomicAdd(&sumsq[col], q);
}

__global__ void k_bn_apply_relu(float* __restrict__ x, const float* __restrict__ sums,
                                const float* __restrict__ sumsq,
                                const float* __restrict__ g, const float* __restrict__ beta,
                                int M) {
  size_t i = (size_t)blockIdx.x * blockDim.x + threadIdx.x;
  if (i >= (size_t)M * 128) return;
  int col = (int)(i & 127);
  float mean = sums[col] / (float)M;
  float var  = sumsq[col] / (float)M - mean * mean;
  float sc = g[col] * rsqrtf(var + 1e-5f);
  float sh = beta[col] - mean * sc;
  x[i] = fmaxf(x[i] * sc + sh, 0.f);
}

// ---------------- GAT edge kernels (HEADS=4, OUTH=32) ----------------

// el[i,h] = sum_c feat[i,h,c]*al[h,c]; er likewise
__global__ void k_el_er(const float* __restrict__ feat, const float* __restrict__ al,
                        const float* __restrict__ ar, float* __restrict__ el,
                        float* __restrict__ er, int M) {
  int idx = blockIdx.x * blockDim.x + threadIdx.x;
  if (idx >= M * 4) return;
  int i = idx >> 2, h = idx & 3;
  const float* f = feat + (size_t)i * 128 + h * 32;
  float sl = 0.f, sr = 0.f;
#pragma unroll
  for (int c = 0; c < 32; ++c) { sl += f[c] * al[h * 32 + c]; sr += f[c] * ar[h * 32 + c]; }
  el[idx] = sl; er[idx] = sr;
}

__global__ void k_gat_edge1(const int* __restrict__ src, const int* __restrict__ dst,
                            const float* __restrict__ el, const float* __restrict__ er,
                            float* __restrict__ eb, float* __restrict__ mmax, int E) {
  int e = blockIdx.x * blockDim.x + threadIdx.x;
  if (e >= E) return;
  int s = src[e], d = dst[e];
#pragma unroll
  for (int h = 0; h < 4; ++h) {
    float v = el[s * 4 + h] + er[d * 4 + h];
    v = (v > 0.f) ? v : 0.2f * v;        // LeakyReLU(0.2)
    eb[(size_t)e * 4 + h] = v;
    atomicMaxFloat(&mmax[d * 4 + h], v);
  }
}

__global__ void k_gat_edge2(const int* __restrict__ dst, float* __restrict__ eb,
                            const float* __restrict__ mmax, float* __restrict__ den, int E) {
  int e = blockIdx.x * blockDim.x + threadIdx.x;
  if (e >= E) return;
  int d = dst[e];
#pragma unroll
  for (int h = 0; h < 4; ++h) {
    float a = expf(eb[(size_t)e * 4 + h] - mmax[d * 4 + h]);
    eb[(size_t)e * 4 + h] = a;
    atomicAdd(&den[d * 4 + h], a);
  }
}

__global__ void k_gat_edge3(const int* __restrict__ src, const int* __restrict__ dst,
                            const float* __restrict__ feat, const float* __restrict__ eb,
                            const float* __restrict__ den, float* __restrict__ out, int E) {
  size_t idx = (size_t)blockIdx.x * blockDim.x + threadIdx.x;
  if (idx >= (size_t)E * 128) return;
  int e = (int)(idx >> 7), c = (int)(idx & 127), h = c >> 5;
  int s = src[e], d = dst[e];
  float w = eb[(size_t)e * 4 + h] / den[d * 4 + h];  // den>0 wherever an edge exists
  atomicAdd(&out[(size_t)d * 128 + c], feat[(size_t)s * 128 + c] * w);
}

// ---------------- generic edge scatter kernels ----------------

__global__ void k_edge_add(const int* __restrict__ src, const int* __restrict__ dst,
                           const float* __restrict__ x, float* __restrict__ agg,
                           int E, int C) {
  size_t idx = (size_t)blockIdx.x * blockDim.x + threadIdx.x;
  if (idx >= (size_t)E * C) return;
  int e = (int)(idx / C), c = (int)(idx % C);
  atomicAdd(&agg[(size_t)dst[e] * C + c], x[(size_t)src[e] * C + c]);
}

__global__ void k_edge_max(const int* __restrict__ src, const int* __restrict__ dst,
                           const float* __restrict__ x, float* __restrict__ out,
                           int E, int C) {
  size_t idx = (size_t)blockIdx.x * blockDim.x + threadIdx.x;
  if (idx >= (size_t)E * C) return;
  int e = (int)(idx / C), c = (int)(idx % C);
  atomicMaxFloat(&out[(size_t)dst[e] * C + c], x[(size_t)src[e] * C + c]);
}

__global__ void k_readout(const float* __restrict__ x, const int* __restrict__ gid,
                          float* __restrict__ out, int M, int C) {
  size_t idx = (size_t)blockIdx.x * blockDim.x + threadIdx.x;
  if (idx >= (size_t)M * C) return;
  int i = (int)(idx / C), c = (int)(idx % C);
  atomicAdd(&out[(size_t)gid[i] * C + c], x[idx]);
}

// ---------------- tiny output MLP (B=8) ----------------

__global__ void k_out_mlp1(const float* __restrict__ h1, const float* __restrict__ h2,
                           const float* __restrict__ W1, const float* __restrict__ b1,
                           float* __restrict__ hmid) {
  int idx = blockIdx.x * blockDim.x + threadIdx.x;  // 8*128
  if (idx >= 8 * 128) return;
  int row = idx >> 7, col = idx & 127;
  float acc = b1[col];
  for (int k = 0; k < 256; ++k) acc += h1[row * 256 + k] * W1[k * 128 + col];
  for (int k = 0; k < 128; ++k) acc += h2[row * 128 + k] * W1[(256 + k) * 128 + col];
  hmid[idx] = fmaxf(acc, 0.f);
}

__global__ void k_out_mlp2(const float* __restrict__ hmid, const float* __restrict__ W2,
                           const float* __restrict__ b2, float* __restrict__ out) {
  int idx = blockIdx.x * blockDim.x + threadIdx.x;  // 8*16
  if (idx >= 8 * 16) return;
  int row = idx >> 4, col = idx & 15;
  float acc = b2[col];
  for (int k = 0; k < 128; ++k) acc += hmid[row * 128 + k] * W2[k * 16 + col];
  out[idx] = acc;
}

// ---------------- host orchestration ----------------

static inline unsigned cdiv(size_t n, unsigned b) { return (unsigned)((n + b - 1) / b); }

extern "C" void kernel_launch(void* const* d_in, const int* in_sizes, int n_in,
                              void* d_out, int out_size, void* d_ws, size_t ws_size,
                              hipStream_t stream) {
  // --- input mapping (setup_inputs dict order; nested pytrees in sorted-key leaf order) ---
  const float* feats = (const float*)d_in[0];
  // bot_params[t]: base 1+t*10 : Wgat, al, ar, W1, W2, b1, b2, bta, eps, g
  const float *Wgat[2], *al[2], *ar[2], *gW1[2], *gW2[2], *gb1[2], *gb2[2], *gbta[2], *geps[2], *gg[2];
  for (int t = 0; t < 2; ++t) {
    int b = 1 + t * 10;
    Wgat[t] = (const float*)d_in[b + 0]; al[t]  = (const float*)d_in[b + 1];
    ar[t]   = (const float*)d_in[b + 2]; gW1[t] = (const float*)d_in[b + 3];
    gW2[t]  = (const float*)d_in[b + 4]; gb1[t] = (const float*)d_in[b + 5];
    gb2[t]  = (const float*)d_in[b + 6]; gbta[t]= (const float*)d_in[b + 7];
    geps[t] = (const float*)d_in[b + 8]; gg[t]  = (const float*)d_in[b + 9];
  }
  // h2_params[l]: base 21+l*7 : W1, W2, b1, b2, bta, eps, g
  const float *hW1[2], *hW2[2], *hb1[2], *hb2[2], *hbta[2], *heps[2], *hg[2];
  for (int l = 0; l < 2; ++l) {
    int b = 21 + l * 7;
    hW1[l] = (const float*)d_in[b + 0]; hW2[l] = (const float*)d_in[b + 1];
    hb1[l] = (const float*)d_in[b + 2]; hb2[l] = (const float*)d_in[b + 3];
    hbta[l]= (const float*)d_in[b + 4]; heps[l]= (const float*)d_in[b + 5];
    hg[l]  = (const float*)d_in[b + 6];
  }
  const float* oW1 = (const float*)d_in[35];
  const float* oW2 = (const float*)d_in[36];
  const float* ob1 = (const float*)d_in[37];
  const float* ob2 = (const float*)d_in[38];
  const int* src_b  = (const int*)d_in[39];
  const int* dst_b  = (const int*)d_in[40];
  const int* src_g1 = (const int*)d_in[41];
  const int* dst_g1 = (const int*)d_in[42];
  const int* src_i2 = (const int*)d_in[43];
  const int* dst_i2 = (const int*)d_in[44];
  const int* gid_a  = (const int*)d_in[45];
  const int* gid_c2 = (const int*)d_in[46];

  const int NA  = in_sizes[0] / 128;     // 100000
  const int EB  = in_sizes[39] / 2;      // 1000000
  const int EG1 = in_sizes[41];          // 200000
  const int EI2 = in_sizes[43];          // 160000
  const int NC2 = in_sizes[46];          // 10000

  // --- workspace layout (floats) ---
  float* ws = (float*)d_ws;
  size_t o = 0;
  float* featbuf = ws + o; o += (size_t)NA * 128;   // GAT fc out; reused as MLP intermediate
  float* gatout  = ws + o; o += (size_t)NA * 128;
  float* aggbuf  = ws + o; o += (size_t)NA * 128;
  float* hA      = ws + o; o += (size_t)NA * 256;
  float* eb      = ws + o; o += (size_t)EB * 4;
  float* elb     = ws + o; o += (size_t)NA * 4;
  float* erb     = ws + o; o += (size_t)NA * 4;
  float* mmax    = ws + o; o += (size_t)NA * 4;
  float* den     = ws + o; o += (size_t)NA * 4;
  float* hc      = ws + o; o += (size_t)NC2 * 256;  // later reused as final [NC2,128]
  float* hc_agg  = ws + o; o += (size_t)NC2 * 256;
  float* hc_t    = ws + o; o += (size_t)NC2 * 128;
  float* hc2     = ws + o; o += (size_t)NC2 * 128;
  float* sums    = ws + o; o += 128;
  float* sumsq   = ws + o; o += 128;
  float* h1      = ws + o; o += 8 * 256;
  float* h2s     = ws + o; o += 8 * 128;
  float* hmid    = ws + o; o += 8 * 128;
  __bf16* wt = (__bf16*)(ws + o);
  __bf16* wt_gat[2] = { wt, wt + 16384 };
  __bf16* wt_g1[2]  = { wt + 2 * 16384, wt + 3 * 16384 };
  __bf16* wt_g2[2]  = { wt + 4 * 16384, wt + 5 * 16384 };
  __bf16* wt_h1a    = wt + 6 * 16384;          // 256x128 -> 32768
  __bf16* wt_h2a    = wt + 6 * 16384 + 32768;
  __bf16* wt_h1b    = wt_h2a + 16384;
  __bf16* wt_h2b    = wt_h1b + 16384;

  const unsigned T = 256;
  // --- transpose all WMMA weights to bf16 [N,K] ---
  for (int t = 0; t < 2; ++t) {
    k_transpose_bf16<<<cdiv(16384, T), T, 0, stream>>>(Wgat[t], wt_gat[t], 128, 128);
    k_transpose_bf16<<<cdiv(16384, T), T, 0, stream>>>(gW1[t], wt_g1[t], 128, 128);
    k_transpose_bf16<<<cdiv(16384, T), T, 0, stream>>>(gW2[t], wt_g2[t], 128, 128);
  }
  k_transpose_bf16<<<cdiv(32768, T), T, 0, stream>>>(hW1[0], wt_h1a, 256, 128);
  k_transpose_bf16<<<cdiv(16384, T), T, 0, stream>>>(hW2[0], wt_h2a, 128, 128);
  k_transpose_bf16<<<cdiv(16384, T), T, 0, stream>>>(hW1[1], wt_h1b, 128, 128);
  k_transpose_bf16<<<cdiv(16384, T), T, 0, stream>>>(hW2[1], wt_h2b, 128, 128);

  const float NEG_INF = -__builtin_inff();
  const size_t NA128 = (size_t)NA * 128;

  for (int t = 0; t < 2; ++t) {
    const int* sb = src_b + (size_t)t * EB;
    const int* db = dst_b + (size_t)t * EB;
    // GAT: fc
    k_gemm_wmma<<<cdiv((size_t)NA, 128), 256, 0, stream>>>(feats, 128, wt_gat[t], nullptr,
                                                           featbuf, 128, NA, 128, 0);
    k_el_er<<<cdiv((size_t)NA * 4, T), T, 0, stream>>>(featbuf, al[t], ar[t], elb, erb, NA);
    k_fill<<<cdiv((size_t)NA * 4, T), T, 0, stream>>>(mmax, NEG_INF, (size_t)NA * 4);
    k_fill<<<cdiv((size_t)NA * 4, T), T, 0, stream>>>(den, 0.f, (size_t)NA * 4);
    k_fill<<<cdiv(NA128, T), T, 0, stream>>>(gatout, 0.f, NA128);
    k_gat_edge1<<<cdiv((size_t)EB, T), T, 0, stream>>>(sb, db, elb, erb, eb, mmax, EB);
    k_fix_neginf<<<cdiv((size_t)NA * 4, T), T, 0, stream>>>(mmax, (size_t)NA * 4);
    k_gat_edge2<<<cdiv((size_t)EB, T), T, 0, stream>>>(db, eb, mmax, den, EB);
    k_gat_edge3<<<cdiv((size_t)EB * 128, T), T, 0, stream>>>(sb, db, featbuf, eb, den, gatout, EB);
    k_relu<<<cdiv(NA128, T), T, 0, stream>>>(gatout, NA128);
    // GIN: agg = (1+eps)*h + sum_neigh(h)
    k_scale_copy<<<cdiv(NA128, T), T, 0, stream>>>(aggbuf, gatout, geps[t], NA128);
    k_edge_add<<<cdiv((size_t)EB * 128, T), T, 0, stream>>>(sb, db, gatout, aggbuf, EB, 128);
    // MLP layer1 + BN + relu  (featbuf reused as intermediate)
    k_gemm_wmma<<<cdiv((size_t)NA, 128), 256, 0, stream>>>(aggbuf, 128, wt_g1[t], gb1[t],
                                                           featbuf, 128, NA, 128, 0);
    k_fill<<<1, 256, 0, stream>>>(sums, 0.f, 128);
    k_fill<<<1, 256, 0, stream>>>(sumsq, 0.f, 128);
    k_bn_stats<<<512, 128, 0, stream>>>(featbuf, sums, sumsq, NA);
    k_bn_apply_relu<<<cdiv(NA128, T), T, 0, stream>>>(featbuf, sums, sumsq, gg[t], gbta[t], NA);
    // MLP layer2 + relu -> hA column block t
    k_gemm_wmma<<<cdiv((size_t)NA, 128), 256, 0, stream>>>(featbuf, 128, wt_g2[t], gb2[t],
                                                           hA + t * 128, 256, NA, 128, 1);
  }

  // G1: max-reduce hA -> hc [NC2,256]
  k_fill<<<cdiv((size_t)NC2 * 256, T), T, 0, stream>>>(hc, NEG_INF, (size_t)NC2 * 256);
  k_edge_max<<<cdiv((size_t)EG1 * 256, T), T, 0, stream>>>(src_g1, dst_g1, hA, hc, EG1, 256);
  k_fix_neginf<<<cdiv((size_t)NC2 * 256, T), T, 0, stream>>>(hc, (size_t)NC2 * 256);

  // I2 GIN layer 1 (K=256)
  k_scale_copy<<<cdiv((size_t)NC2 * 256, T), T, 0, stream>>>(hc_agg, hc, heps[0], (size_t)NC2 * 256);
  k_edge_add<<<cdiv((size_t)EI2 * 256, T), T, 0, stream>>>(src_i2, dst_i2, hc, hc_agg, EI2, 256);
  k_gemm_wmma<<<cdiv((size_t)NC2, 128), 256, 0, stream>>>(hc_agg, 256, wt_h1a, hb1[0],
                                                          hc_t, 128, NC2, 256, 0);
  k_fill<<<1, 256, 0, stream>>>(sums, 0.f, 128);
  k_fill<<<1, 256, 0, stream>>>(sumsq, 0.f, 128);
  k_bn_stats<<<256, 128, 0, stream>>>(hc_t, sums, sumsq, NC2);
  k_bn_apply_relu<<<cdiv((size_t)NC2 * 128, T), T, 0, stream>>>(hc_t, sums, sumsq, hg[0], hbta[0], NC2);
  k_gemm_wmma<<<cdiv((size_t)NC2, 128), 256, 0, stream>>>(hc_t, 128, wt_h2a, hb2[0],
                                                          hc2, 128, NC2, 128, 1);
  // I2 GIN layer 2 (K=128); final result written to hc (reused, stride 128)
  k_scale_copy<<<cdiv((size_t)NC2 * 128, T), T, 0, stream>>>(hc_agg, hc2, heps[1], (size_t)NC2 * 128);
  k_edge_add<<<cdiv((size_t)EI2 * 128, T), T, 0, stream>>>(src_i2, dst_i2, hc2, hc_agg, EI2, 128);
  k_gemm_wmma<<<cdiv((size_t)NC2, 128), 256, 0, stream>>>(hc_agg, 128, wt_h1b, hb1[1],
                                                          hc_t, 128, NC2, 128, 0);
  k_fill<<<1, 256, 0, stream>>>(sums, 0.f, 128);
  k_fill<<<1, 256, 0, stream>>>(sumsq, 0.f, 128);
  k_bn_stats<<<256, 128, 0, stream>>>(hc_t, sums, sumsq, NC2);
  k_bn_apply_relu<<<cdiv((size_t)NC2 * 128, T), T, 0, stream>>>(hc_t, sums, sumsq, hg[1], hbta[1], NC2);
  k_gemm_wmma<<<cdiv((size_t)NC2, 128), 256, 0, stream>>>(hc_t, 128, wt_h2b, hb2[1],
                                                          hc, 128, NC2, 128, 1);

  // readout + output MLP
  k_fill<<<cdiv((size_t)8 * 256, T), T, 0, stream>>>(h1, 0.f, 8 * 256);
  k_fill<<<cdiv((size_t)8 * 128, T), T, 0, stream>>>(h2s, 0.f, 8 * 128);
  k_readout<<<cdiv((size_t)NA * 256, T), T, 0, stream>>>(hA, gid_a, h1, NA, 256);
  k_readout<<<cdiv((size_t)NC2 * 128, T), T, 0, stream>>>(hc, gid_c2, h2s, NC2, 128);
  k_out_mlp1<<<cdiv(8 * 128, T), T, 0, stream>>>(h1, h2s, oW1, ob1, hmid);
  k_out_mlp2<<<1, 128, 0, stream>>>(hmid, oW2, ob2, (float*)d_out);
}
